// RefPhi4FlashMamba_56650618635019
// MI455X (gfx1250) — compile-verified, matrix operand order checked by reading, not run on hardware
//
#include <hip/hip_runtime.h>

// ---------------- problem constants ----------------
#define DMODEL   2048
#define DINNER   4096
#define DSTATE   16
#define DCONV    4
#define DTRANK   128
#define BATCH_   2
#define SEQ_     2048
#define BS_      (BATCH_ * SEQ_)          // 4096 rows
#define NPROJ    (DTRANK + 2 * DSTATE)    // 160

#define TILE_M 128
#define TILE_N 128
#define TILE_K 32

// ---------------- types ----------------
typedef __attribute__((ext_vector_type(8)))  __bf16 bf16x8;
typedef __attribute__((ext_vector_type(16))) __bf16 bf16x16;
typedef __attribute__((ext_vector_type(8)))  float  f32x8;
typedef __attribute__((ext_vector_type(4)))  unsigned int u32x4;
typedef __attribute__((ext_vector_type(8)))  int i32x8;
typedef __attribute__((ext_vector_type(4)))  int i32x4;

#if defined(__has_builtin)
#if __has_builtin(__builtin_amdgcn_tensor_load_to_lds) && __has_builtin(__builtin_amdgcn_s_wait_tensorcnt)
#define HAVE_TDM 1
#endif
#endif
#ifndef HAVE_TDM
#define HAVE_TDM 0
#endif

__device__ __forceinline__ __bf16 f2bf(float f) {
  unsigned int u = __float_as_uint(f);
  unsigned int r = (u + 0x7FFFu + ((u >> 16) & 1u)) >> 16;
  unsigned short s = (unsigned short)r;
  return __builtin_bit_cast(__bf16, s);
}

// ---------------- f32 -> bf16 convert (grid-stride) ----------------
__global__ void k_f2bf(const float* __restrict__ src, __bf16* __restrict__ dst, long n) {
  long i = (long)blockIdx.x * blockDim.x + threadIdx.x;
  long stride = (long)gridDim.x * blockDim.x;
  for (; i < n; i += stride) dst[i] = f2bf(src[i]);
}

#if HAVE_TDM
// LDS byte offset of a __shared__ object (addrspace(3) pointer value)
__device__ __forceinline__ unsigned lds_offset_of(const void* p) {
  typedef __attribute__((address_space(3))) const char as3char;
  return (unsigned)(unsigned long long)(as3char*)p;
}

// Issue a TDM 2D tile load: tile_d1 rows x tile_d0 elems (bf16) from a row-major
// [rows, stride_elems] tensor into LDS (packed rows of tile_d0*2 bytes).
__device__ __forceinline__ void tdm_load_2d(unsigned lds_byte_off, const void* gaddr,
                                            unsigned tensor_d0, unsigned tensor_d1,
                                            unsigned tile_d0, unsigned tile_d1,
                                            unsigned stride_elems) {
  unsigned long long ga = (unsigned long long)gaddr;
  u32x4 g0;
  g0[0] = 1u;                                             // count=1 (valid), user mode
  g0[1] = lds_byte_off;                                   // lds_addr
  g0[2] = (unsigned)(ga & 0xFFFFFFFFull);                 // global_addr[31:0]
  g0[3] = (unsigned)((ga >> 32) & 0x01FFFFFFull)          // global_addr[56:32]
          | 0x80000000u;                                  // type=2 ("image")
  unsigned w0 = (1u << 16);                               // data_size=1 (2 bytes)
  unsigned w1 = (tensor_d0 & 0xFFFFu) << 16;              // tensor_dim0[15:0]
  unsigned w2 = (tensor_d0 >> 16) | ((tensor_d1 & 0xFFFFu) << 16);
  unsigned w3 = (tensor_d1 >> 16) | (tile_d0 << 16);      // tile_dim0
  unsigned w4 = tile_d1;                                  // tile_dim1 (tile_dim2=0)
  unsigned w5 = stride_elems;                             // tensor_dim0_stride[31:0]
  unsigned w6 = 0, w7 = 0;
  i32x8 g1;
  g1[0]=(int)w0; g1[1]=(int)w1; g1[2]=(int)w2; g1[3]=(int)w3;
  g1[4]=(int)w4; g1[5]=(int)w5; g1[6]=(int)w6; g1[7]=(int)w7;
  i32x4 z4 = (i32x4){0,0,0,0};
#if defined(__clang_major__) && (__clang_major__ >= 23)
  i32x8 z8 = (i32x8){0,0,0,0,0,0,0,0};
  __builtin_amdgcn_tensor_load_to_lds(g0, g1, z4, z4, z8, 0);
#else
  __builtin_amdgcn_tensor_load_to_lds(g0, g1, z4, z4, 0);
#endif
}
#else
// Fallback: cooperative global->LDS staging of a rows x 32 bf16 tile
__device__ __forceinline__ void stage_tile(__bf16* __restrict__ dst,
                                           const __bf16* __restrict__ src,
                                           int rows, int maxRows, int ldk, int tid) {
  for (int idx = tid; idx < rows * 4; idx += 256) {
    int r = idx >> 2, c = idx & 3;
    int rr = (r < maxRows) ? r : (maxRows - 1);
    *(bf16x8*)(dst + r * TILE_K + c * 8) = *(const bf16x8*)(src + (size_t)rr * ldk + c * 8);
  }
}
#endif

// ---------------- WMMA bf16 GEMM: C[M,N] = A[M,K] * B[N,K]^T ----------------
// Block = 256 threads = 8 waves; block tile 128x128; wave tile 64x32 (4x2 WMMA tiles).
// A/B tiles staged into double-buffered LDS by the Tensor Data Mover (wave 0 issues,
// TENSORcnt + workgroup barrier publishes); fragments read as ds_load_b128 pairs.
// Fragment layout per CDNA5 ISA 7.12.2: per lane row = lane&15, half = lane>>4;
// chunk0 bf16[8] at k = half*8, chunk1 at k = 16 + half*8 (both 16B-contiguous).
__global__ __launch_bounds__(256)
void k_gemm_bf16(const __bf16* __restrict__ A, const __bf16* __restrict__ B,
                 float* __restrict__ C, int M, int N, int K) {
  __shared__ __bf16 sA[2][TILE_M * TILE_K];   // 8 KB each
  __shared__ __bf16 sB[2][TILE_N * TILE_K];

  const int lane  = threadIdx.x & 31;
  const int wave  = threadIdx.x >> 5;
  const int waveM = wave >> 2;   // 0..1
  const int waveN = wave & 3;    // 0..3
  const int half  = lane >> 4;
  const int l16   = lane & 15;
  const int mBlk  = blockIdx.y * TILE_M;
  const int nBlk  = blockIdx.x * TILE_N;

  f32x8 acc[4][2];
#pragma unroll
  for (int i = 0; i < 4; ++i)
#pragma unroll
    for (int j = 0; j < 2; ++j) acc[i][j] = (f32x8){0.f,0.f,0.f,0.f,0.f,0.f,0.f,0.f};

  const int nIter = K / TILE_K;

#if HAVE_TDM
  if (wave == 0) {
    tdm_load_2d(lds_offset_of(&sA[0][0]), A + (size_t)mBlk * K,
                (unsigned)K, (unsigned)(M - mBlk), TILE_K, TILE_M, (unsigned)K);
    tdm_load_2d(lds_offset_of(&sB[0][0]), B + (size_t)nBlk * K,
                (unsigned)K, (unsigned)(N - nBlk), TILE_K, TILE_N, (unsigned)K);
  }
#else
  stage_tile(&sA[0][0], A + (size_t)mBlk * K, TILE_M, M - mBlk, K, threadIdx.x);
  stage_tile(&sB[0][0], B + (size_t)nBlk * K, TILE_N, N - nBlk, K, threadIdx.x);
#endif

  for (int it = 0; it < nIter; ++it) {
    const int cur = it & 1;
#if HAVE_TDM
    if (wave == 0) __builtin_amdgcn_s_wait_tensorcnt(0);
    __syncthreads();   // tile[cur] ready for all; tile[cur^1] readers (it-1) done
    if (it + 1 < nIter && wave == 0) {
      const int k0n = (it + 1) * TILE_K;
      tdm_load_2d(lds_offset_of(&sA[cur ^ 1][0]), A + (size_t)mBlk * K + k0n,
                  (unsigned)K, (unsigned)(M - mBlk), TILE_K, TILE_M, (unsigned)K);
      tdm_load_2d(lds_offset_of(&sB[cur ^ 1][0]), B + (size_t)nBlk * K + k0n,
                  (unsigned)K, (unsigned)(N - nBlk), TILE_K, TILE_N, (unsigned)K);
    }
#else
    __syncthreads();
    if (it + 1 < nIter) {
      const int k0n = (it + 1) * TILE_K;
      stage_tile(&sA[cur ^ 1][0], A + (size_t)mBlk * K + k0n, TILE_M, M - mBlk, K, threadIdx.x);
      stage_tile(&sB[cur ^ 1][0], B + (size_t)nBlk * K + k0n, TILE_N, N - nBlk, K, threadIdx.x);
    }
#endif

    bf16x16 afrag[4];
#pragma unroll
    for (int i = 0; i < 4; ++i) {
      const __bf16* p = &sA[cur][(waveM * 64 + i * 16 + l16) * TILE_K];
      bf16x8 lo = *(const bf16x8*)(p + half * 8);
      bf16x8 hi = *(const bf16x8*)(p + 16 + half * 8);
      afrag[i] = __builtin_shufflevector(lo, hi, 0,1,2,3,4,5,6,7,8,9,10,11,12,13,14,15);
    }
    bf16x16 bfrag[2];
#pragma unroll
    for (int j = 0; j < 2; ++j) {
      const __bf16* p = &sB[cur][(waveN * 32 + j * 16 + l16) * TILE_K];
      bf16x8 lo = *(const bf16x8*)(p + half * 8);
      bf16x8 hi = *(const bf16x8*)(p + 16 + half * 8);
      bfrag[j] = __builtin_shufflevector(lo, hi, 0,1,2,3,4,5,6,7,8,9,10,11,12,13,14,15);
    }

#pragma unroll
    for (int i = 0; i < 4; ++i)
#pragma unroll
      for (int j = 0; j < 2; ++j)
        acc[i][j] = __builtin_amdgcn_wmma_f32_16x16x32_bf16(
            false, afrag[i], false, bfrag[j], (short)0, acc[i][j], false, false);
  }

  // C/D layout: lane<16 -> rows M=r (r=0..7); lane>=16 -> rows M=8+r; col N = lane&15.
  const int mbase = half * 8;
#pragma unroll
  for (int i = 0; i < 4; ++i) {
#pragma unroll
    for (int j = 0; j < 2; ++j) {
      int col = nBlk + waveN * 32 + j * 16 + l16;
      if (col < N) {
#pragma unroll
        for (int r = 0; r < 8; ++r) {
          int row = mBlk + waveM * 64 + i * 16 + mbase + r;
          if (row < M) C[(size_t)row * N + col] = acc[i][j][r];
        }
      }
    }
  }
}

// ---------------- causal depthwise conv(4) + SiLU ----------------
__global__ void k_conv_silu(const float* __restrict__ xz,
                            const float* __restrict__ conv_w,
                            const float* __restrict__ conv_b,
                            float* __restrict__ xc, __bf16* __restrict__ xcb) {
  long i = (long)blockIdx.x * blockDim.x + threadIdx.x;
  if (i >= (long)BS_ * DINNER) return;
  int  d  = (int)(i & (DINNER - 1));
  long bs = i >> 12;
  int  s  = (int)(bs & (SEQ_ - 1));
  long b  = bs >> 11;
  float acc = conv_b[d];
#pragma unroll
  for (int j = 0; j < DCONV; ++j) {
    int sp = s - (DCONV - 1) + j;
    if (sp >= 0)
      acc += conv_w[d * DCONV + j] * xz[((b * SEQ_ + sp) * (size_t)(2 * DINNER)) + d];
  }
  float sv = acc / (1.f + expf(-acc));
  xc[i]  = sv;
  xcb[i] = f2bf(sv);
}

// ---------------- extract dt_low (proj[:, :128]) as bf16 ----------------
__global__ void k_dtlow(const float* __restrict__ proj, __bf16* __restrict__ dst) {
  long i = (long)blockIdx.x * blockDim.x + threadIdx.x;
  if (i >= (long)BS_ * DTRANK) return;
  long row = i >> 7;
  int  r   = (int)(i & (DTRANK - 1));
  dst[i] = f2bf(proj[row * NPROJ + r]);
}

// ---------------- selective scan ----------------
__global__ __launch_bounds__(256)
void k_scan(const float* __restrict__ xc,
            const float* __restrict__ xz,
            const float* __restrict__ dt,
            const float* __restrict__ proj,
            const float* __restrict__ dt_b,
            const float* __restrict__ A_log,
            const float* __restrict__ Dp,
            __bf16* __restrict__ y_bf) {
  const int tid = threadIdx.x;
  const int b   = blockIdx.x >> 4;
  const int d   = ((blockIdx.x & 15) << 8) + tid;

  float An[DSTATE];
#pragma unroll
  for (int n = 0; n < DSTATE; ++n) An[n] = -expf(A_log[d * DSTATE + n]);
  float st[DSTATE];
#pragma unroll
  for (int n = 0; n < DSTATE; ++n) st[n] = 0.f;
  const float dtb = dt_b[d];
  const float Dd  = Dp[d];

  __shared__ float sBC[2 * DSTATE];

  for (int s = 0; s < SEQ_; ++s) {
    const long bs = (long)b * SEQ_ + s;
    __syncthreads();
    if (tid < 2 * DSTATE) sBC[tid] = proj[bs * NPROJ + DTRANK + tid];
    __syncthreads();

    float dtv = dt[bs * DINNER + d] + dtb;
    dtv = (dtv > 20.f) ? dtv : log1pf(expf(dtv));
    const float xv  = xc[bs * DINNER + d];
    const float zv  = xz[bs * (size_t)(2 * DINNER) + DINNER + d];
    const float xdt = xv * dtv;
    float y = 0.f;
#pragma unroll
    for (int n = 0; n < DSTATE; ++n) {
      float dA = expf(dtv * An[n]);
      st[n] = st[n] * dA + xdt * sBC[n];
      y += st[n] * sBC[DSTATE + n];
    }
    y += Dd * xv;
    y *= zv / (1.f + expf(-zv));
    y_bf[bs * DINNER + d] = f2bf(y);
  }
}

// ---------------- host-side launch ----------------
static inline size_t alignup(size_t x) { return (x + 255) & ~(size_t)255; }

extern "C" void kernel_launch(void* const* d_in, const int* in_sizes, int n_in,
                              void* d_out, int out_size, void* d_ws, size_t ws_size,
                              hipStream_t stream) {
  (void)in_sizes; (void)n_in; (void)out_size; (void)ws_size;
  const float* hs     = (const float*)d_in[0];
  const float* W_in   = (const float*)d_in[1];
  const float* conv_w = (const float*)d_in[2];
  const float* conv_b = (const float*)d_in[3];
  const float* W_x    = (const float*)d_in[4];
  const float* dt_w   = (const float*)d_in[5];
  const float* dt_b   = (const float*)d_in[6];
  const float* A_log  = (const float*)d_in[7];
  const float* Dp     = (const float*)d_in[8];
  const float* W_out  = (const float*)d_in[9];
  float* out = (float*)d_out;

  char* ws = (char*)d_ws;
  size_t off = 0;
  auto take = [&](size_t bytes) { char* p = ws + off; off = alignup(off + bytes); return p; };
  __bf16* hsb    = (__bf16*)take((size_t)BS_ * DMODEL * 2);
  __bf16* winb   = (__bf16*)take((size_t)2 * DINNER * DMODEL * 2);
  float*  xzf    = (float*) take((size_t)BS_ * 2 * DINNER * 4);
  float*  xc     = (float*) take((size_t)BS_ * DINNER * 4);
  __bf16* xcb    = (__bf16*)take((size_t)BS_ * DINNER * 2);
  __bf16* wxb    = (__bf16*)take((size_t)NPROJ * DINNER * 2);
  float*  proj   = (float*) take((size_t)BS_ * NPROJ * 4);
  __bf16* dtlowb = (__bf16*)take((size_t)BS_ * DTRANK * 2);
  __bf16* dtwb   = (__bf16*)take((size_t)DINNER * DTRANK * 2);
  float*  dtf    = (float*) take((size_t)BS_ * DINNER * 4);
  __bf16* yb     = (__bf16*)take((size_t)BS_ * DINNER * 2);
  __bf16* woutb  = (__bf16*)take((size_t)DMODEL * DINNER * 2);

  const int CT = 256;
  k_f2bf<<<4096, CT, 0, stream>>>(hs,    hsb,   (long)BS_ * DMODEL);
  k_f2bf<<<4096, CT, 0, stream>>>(W_in,  winb,  (long)2 * DINNER * DMODEL);
  k_f2bf<<<1024, CT, 0, stream>>>(W_x,   wxb,   (long)NPROJ * DINNER);
  k_f2bf<<<1024, CT, 0, stream>>>(dt_w,  dtwb,  (long)DINNER * DTRANK);
  k_f2bf<<<4096, CT, 0, stream>>>(W_out, woutb, (long)DMODEL * DINNER);

  {
    dim3 grid((2 * DINNER + TILE_N - 1) / TILE_N, (BS_ + TILE_M - 1) / TILE_M);
    k_gemm_bf16<<<grid, CT, 0, stream>>>(hsb, winb, xzf, BS_, 2 * DINNER, DMODEL);
  }
  {
    long n = (long)BS_ * DINNER;
    k_conv_silu<<<(unsigned)((n + CT - 1) / CT), CT, 0, stream>>>(xzf, conv_w, conv_b, xc, xcb);
  }
  {
    dim3 grid((NPROJ + TILE_N - 1) / TILE_N, (BS_ + TILE_M - 1) / TILE_M);
    k_gemm_bf16<<<grid, CT, 0, stream>>>(xcb, wxb, proj, BS_, NPROJ, DINNER);
  }
  {
    long n = (long)BS_ * DTRANK;
    k_dtlow<<<(unsigned)((n + CT - 1) / CT), CT, 0, stream>>>(proj, dtlowb);
  }
  {
    dim3 grid((DINNER + TILE_N - 1) / TILE_N, (BS_ + TILE_M - 1) / TILE_M);
    k_gemm_bf16<<<grid, CT, 0, stream>>>(dtlowb, dtwb, dtf, BS_, DINNER, DTRANK);
  }
  k_scan<<<BATCH_ * (DINNER / 256), CT, 0, stream>>>(xc, xzf, dtf, proj, dt_b, A_log, Dp, yb);
  {
    dim3 grid((DMODEL + TILE_N - 1) / TILE_N, (BS_ + TILE_M - 1) / TILE_M);
    k_gemm_bf16<<<grid, CT, 0, stream>>>(yb, woutb, out, BS_, DMODEL, DINNER);
  }
}
